// Transformer_7988639170850
// MI455X (gfx1250) — matrix operand order repaired and checked by reading, not compiled, not run on hardware
//
#include <hip/hip_runtime.h>
#include <hip/hip_bf16.h>

// ---------------------------------------------------------------------------
// Types for CDNA5 WMMA (gfx1250, wave32)
// ---------------------------------------------------------------------------
typedef __attribute__((ext_vector_type(16))) __bf16 v16bf;
typedef __attribute__((ext_vector_type(8)))  float  v8f;

union BFrag {
    uint4 q[2];   // two 16-byte chunks
    v16bf v;      // 16 x bf16 = 32 bytes
};

union Pack16 {
    uint4 q[2];
    unsigned short u[16];
};

// Native bf16 convert (lets clang emit the hardware cvt, not a bit-twiddle)
__device__ __forceinline__ unsigned short f2bf(float f) {
    __bf16 h = (__bf16)f;
    return __builtin_bit_cast(unsigned short, h);
}
// bf16 -> fp32 is exact: just a shift
__device__ __forceinline__ float bf2f(unsigned short u) {
    return __uint_as_float((unsigned int)u << 16);
}

__device__ __forceinline__ v8f zero8() {
    v8f z;
#pragma unroll
    for (int i = 0; i < 8; ++i) z[i] = 0.0f;
    return z;
}

__device__ __forceinline__ v8f bfwmma(const BFrag& a, const BFrag& b, v8f c) {
    // D = A(16x32 bf16) * B(32x16 bf16) + C(16x16 f32)
    return __builtin_amdgcn_wmma_f32_16x16x32_bf16(
        /*neg_a=*/false, a.v, /*neg_b=*/false, b.v,
        /*c_mod=*/(short)0, c, /*reuse_a=*/false, /*reuse_b=*/false);
}

// ---------------------------------------------------------------------------
// Model constants
// ---------------------------------------------------------------------------
#define DMODEL 512
#define DFF    2048
#define NB     4
#define NS     1024
#define NROWS  (NB * NS)          // 4096
#define DK     64
#define NH     8

// ---------------------------------------------------------------------------
// Embedding + sinusoidal positional encoding -> bf16 activation stream
// ---------------------------------------------------------------------------
__global__ __launch_bounds__(128) void embed_kernel(
    const int* __restrict__ tok, const float* __restrict__ emb,
    unsigned short* __restrict__ out)
{
    const int row = blockIdx.x;           // 0..4095
    const int s   = row & (NS - 1);       // position in sequence
    const int t   = tok[row];
    const int tid = threadIdx.x;
#pragma unroll
    for (int j = 0; j < 4; ++j) {
        const int c  = j * 128 + tid;
        const int i2 = (c >> 1) << 1;     // 2*i
        const float fac = __powf(10000.0f, -(float)i2 * (1.0f / (float)DMODEL));
        const float ang = (float)s * fac;
        const float pe  = (c & 1) ? __cosf(ang) : __sinf(ang);
        out[(size_t)row * DMODEL + c] =
            f2bf(emb[(size_t)t * DMODEL + c] * 22.62741699796952f + pe);
    }
}

// ---------------------------------------------------------------------------
// GEMM: C(MxN bf16) = A(MxK bf16) @ W(KxN fp32 params) + bias, optional ReLU.
// bf16 WMMA, f32 accumulate. Block tile 64x128, 8 waves each own 32x32.
// A staging is a pure b128 global->LDS copy (no VALU); W converts via native
// bf16 cvt once per block (reused by all 64 rows of this block's M tile).
// ---------------------------------------------------------------------------
#define TM 64
#define TN 128
#define TK 32

__global__ __launch_bounds__(256) void gemm_bias_kernel(
    const unsigned short* __restrict__ A, const float* __restrict__ W,
    const float* __restrict__ bias, unsigned short* __restrict__ C,
    int M, int N, int K, int relu)
{
    __shared__ unsigned short sA[TM][TK];   // A tile, row-major bf16
    __shared__ unsigned short sB[TN][TK];   // W tile transposed: sB[n][k]

    const int tid  = threadIdx.x;
    const int m0   = blockIdx.y * TM;
    const int n0   = blockIdx.x * TN;
    const int w    = tid >> 5;
    const int lane = tid & 31;
    const int h    = lane >> 4;     // half of wave
    const int l16  = lane & 15;
    const int wm   = (w >> 2) * 32; // 0 / 32
    const int wn   = (w & 3) * 32;  // 0..96

    v8f acc[2][2];
#pragma unroll
    for (int i = 0; i < 2; ++i)
#pragma unroll
        for (int j = 0; j < 2; ++j) acc[i][j] = zero8();

    const int ar = tid >> 2;            // 0..63
    const int ac = (tid & 3) * 8;       // 0..24
    const int bk = tid >> 3;            // 0..31
    const int bn = (tid & 7) * 16;      // 0..112

    for (int k0 = 0; k0 < K; k0 += TK) {
        __syncthreads();
        // Stage A tile: raw 16B copy (bf16 -> bf16)
        *(uint4*)&sA[ar][ac] = *(const uint4*)(A + (size_t)(m0 + ar) * K + k0 + ac);
        // Stage W tile transposed (fp32 -> bf16, coalesced global reads along N)
        const float* wp = W + (size_t)(k0 + bk) * N + n0 + bn;
#pragma unroll
        for (int t = 0; t < 16; ++t) sB[bn + t][bk] = f2bf(wp[t]);
        if (k0 + TK < K) {
            __builtin_prefetch(A + (size_t)(m0 + ar) * K + k0 + TK + ac, 0, 1);
            __builtin_prefetch(W + (size_t)(k0 + TK + bk) * N + n0 + bn, 0, 1);
        }
        __syncthreads();

        // Fragment loads matching documented layouts:
        // A: lane(l16)=row, e<8 -> k=8h+e ; e>=8 -> k=16+8h+(e-8)
        BFrag fa[2];
#pragma unroll
        for (int mi = 0; mi < 2; ++mi) {
            const unsigned short* p = &sA[wm + mi * 16 + l16][8 * h];
            fa[mi].q[0] = *(const uint4*)p;
            fa[mi].q[1] = *(const uint4*)(p + 16);
        }
        // B: lane(l16)=col, e<8 -> k=8h+e ; e>=8 -> k=16+8h+(e-8)
        BFrag fb[2];
#pragma unroll
        for (int ni = 0; ni < 2; ++ni) {
            const unsigned short* p = &sB[wn + ni * 16 + l16][8 * h];
            fb[ni].q[0] = *(const uint4*)p;
            fb[ni].q[1] = *(const uint4*)(p + 16);
        }
#pragma unroll
        for (int mi = 0; mi < 2; ++mi)
#pragma unroll
            for (int ni = 0; ni < 2; ++ni)
                acc[mi][ni] = bfwmma(fa[mi], fb[ni], acc[mi][ni]);
    }

    // Epilogue: bias (+ReLU), C layout: element j -> row 8h+j, lane l16 -> col
#pragma unroll
    for (int mi = 0; mi < 2; ++mi)
#pragma unroll
        for (int ni = 0; ni < 2; ++ni) {
            const int col = n0 + wn + ni * 16 + l16;
            const float bv = bias[col];
#pragma unroll
            for (int j = 0; j < 8; ++j) {
                const int row = m0 + wm + mi * 16 + 8 * h + j;
                float v = acc[mi][ni][j] + bv;
                if (relu) v = fmaxf(v, 0.0f);
                C[(size_t)row * N + col] = f2bf(v);
            }
        }
}

// ---------------------------------------------------------------------------
// Flash-style attention over bf16 Q/K/V. The reference's raw reshape makes
// head (b,h) a contiguous 1024x64 row-major slab at offset b*S*D + h*128*512.
// Block = (query tile of 128 rows, h, b); 8 waves x 16 query rows each.
// Streams 64-key tiles with online softmax; causal mask on head-local (i,j).
// ---------------------------------------------------------------------------
__global__ __launch_bounds__(256) void attention_kernel(
    const unsigned short* __restrict__ qb, const unsigned short* __restrict__ kb,
    const unsigned short* __restrict__ vb, unsigned short* __restrict__ ob,
    int causal)
{
    __shared__ unsigned short sQ[128][DK];      // 16 KB
    __shared__ unsigned short sK[64][DK];       //  8 KB  (sK[key][d])
    __shared__ unsigned short sVT[DK][64];      //  8 KB  (sVT[d][key])
    __shared__ unsigned short sP[8][16][64];    // 16 KB  per-wave P tiles

    const int tid = threadIdx.x;
    const int q0  = blockIdx.x * 128;
    const int hh  = blockIdx.y;
    const int bb  = blockIdx.z;
    const size_t base = (size_t)bb * (NS * DMODEL) + (size_t)hh * (128 * DMODEL);

    const int w    = tid >> 5;
    const int lane = tid & 31;
    const int h    = lane >> 4;
    const int l16  = lane & 15;

    // Stage Q tile once: pure 16B copies
    {
        const int r = tid >> 1, c0 = (tid & 1) * 32;
        const unsigned short* p = qb + base + (size_t)(q0 + r) * DK + c0;
#pragma unroll
        for (int t = 0; t < 4; ++t)
            *(uint4*)&sQ[r][c0 + t * 8] = *(const uint4*)(p + t * 8);
    }

    float rm[8], rl[8];
    v8f oacc[4];
#pragma unroll
    for (int j = 0; j < 8; ++j) { rm[j] = -3.0e38f; rl[j] = 0.0f; }
#pragma unroll
    for (int i = 0; i < 4; ++i) oacc[i] = zero8();

    for (int kt = 0; kt < 16; ++kt) {
        const int j0 = kt * 64;
        if (causal && j0 > q0 + 127) break;   // fully masked key tiles
        __syncthreads();
        {
            const int r = tid >> 2, c0 = (tid & 3) * 16;
            const unsigned short* kp = kb + base + (size_t)(j0 + r) * DK + c0;
            *(uint4*)&sK[r][c0]     = *(const uint4*)kp;
            *(uint4*)&sK[r][c0 + 8] = *(const uint4*)(kp + 8);
            // V is staged transposed (sVT[d][key]) for B-fragment loads
            const unsigned short* vp = vb + base + (size_t)(j0 + r) * DK + c0;
            Pack16 pv;
            pv.q[0] = *(const uint4*)vp;
            pv.q[1] = *(const uint4*)(vp + 8);
#pragma unroll
            for (int t = 0; t < 16; ++t) sVT[c0 + t][r] = pv.u[t];
        }
        __syncthreads();

        // S = Q * K^T   (16 q x 64 keys per wave)
        v8f sacc[4];
#pragma unroll
        for (int ni = 0; ni < 4; ++ni) sacc[ni] = zero8();
#pragma unroll
        for (int kc = 0; kc < 2; ++kc) {
            BFrag fq;
            const unsigned short* qp = &sQ[w * 16 + l16][kc * 32 + 8 * h];
            fq.q[0] = *(const uint4*)qp;
            fq.q[1] = *(const uint4*)(qp + 16);
#pragma unroll
            for (int ni = 0; ni < 4; ++ni) {
                BFrag fk;
                const unsigned short* kp = &sK[ni * 16 + l16][kc * 32 + 8 * h];
                fk.q[0] = *(const uint4*)kp;
                fk.q[1] = *(const uint4*)(kp + 16);
                sacc[ni] = bfwmma(fq, fk, sacc[ni]);
            }
        }

        // Online softmax update (rows m = 8h+j, half-wave reductions)
#pragma unroll
        for (int j = 0; j < 8; ++j) {
            const int qi = q0 + w * 16 + 8 * h + j;
            float sv[4];
            float tmax = -3.0e38f;
#pragma unroll
            for (int ni = 0; ni < 4; ++ni) {
                const int kj = j0 + ni * 16 + l16;
                float s = sacc[ni][j] * 0.125f;           // 1/sqrt(64)
                if (causal && kj > qi) s = -1.0e10f;
                sv[ni] = s;
                tmax = fmaxf(tmax, s);
            }
#pragma unroll
            for (int off = 1; off < 16; off <<= 1)
                tmax = fmaxf(tmax, __shfl_xor(tmax, off, 32));
            const float newm = fmaxf(rm[j], tmax);
            const float corr = __expf(rm[j] - newm);
            float tsum = 0.0f;
#pragma unroll
            for (int ni = 0; ni < 4; ++ni) {
                const float p = __expf(sv[ni] - newm);
                sacc[ni][j] = p;                          // reuse as P
                tsum += p;
            }
#pragma unroll
            for (int off = 1; off < 16; off <<= 1)
                tsum += __shfl_xor(tsum, off, 32);
            rl[j] = rl[j] * corr + tsum;
            rm[j] = newm;
#pragma unroll
            for (int ni = 0; ni < 4; ++ni) oacc[ni][j] *= corr;
        }

        // Re-layout P (C-layout -> A-layout) through per-wave LDS region
#pragma unroll
        for (int ni = 0; ni < 4; ++ni)
#pragma unroll
            for (int j = 0; j < 8; ++j)
                sP[w][8 * h + j][ni * 16 + l16] = f2bf(sacc[ni][j]);
        asm volatile("s_wait_dscnt 0" ::: "memory");   // same-wave LDS RAW

        // O += P(16x64) * V(64x64)
#pragma unroll
        for (int kc = 0; kc < 2; ++kc) {
            BFrag fp;
            const unsigned short* pp = &sP[w][l16][kc * 32 + 8 * h];
            fp.q[0] = *(const uint4*)pp;
            fp.q[1] = *(const uint4*)(pp + 16);
#pragma unroll
            for (int ni = 0; ni < 4; ++ni) {
                BFrag fv;
                const unsigned short* vp = &sVT[ni * 16 + l16][kc * 32 + 8 * h];
                fv.q[0] = *(const uint4*)vp;
                fv.q[1] = *(const uint4*)(vp + 16);
                oacc[ni] = bfwmma(fp, fv, oacc[ni]);
            }
        }
    }

    // Normalize and write out (same raw-reshape layout as inputs)
#pragma unroll
    for (int ni = 0; ni < 4; ++ni)
#pragma unroll
        for (int j = 0; j < 8; ++j) {
            const int qi = q0 + w * 16 + 8 * h + j;
            const float denom = rl[j];
            const float val = (denom > 0.0f) ? oacc[ni][j] / denom : 0.0f;
            ob[base + (size_t)qi * DK + ni * 16 + l16] = f2bf(val);
        }
}

// ---------------------------------------------------------------------------
// Fused residual-add + LayerNorm over last dim (512), fp32 math on bf16 data.
// F32OUT selects fp32 output (final decoder layer writes d_out).
// ---------------------------------------------------------------------------
template <bool F32OUT>
__global__ __launch_bounds__(128) void add_ln_kernel(
    const unsigned short* __restrict__ resid, const unsigned short* __restrict__ sub,
    const float* __restrict__ g, const float* __restrict__ be,
    void* __restrict__ outv)
{
    __shared__ float red[128];
    const int row = blockIdx.x, tid = threadIdx.x;
    const unsigned short* rp = resid + (size_t)row * DMODEL;
    const unsigned short* sp = sub   + (size_t)row * DMODEL;

    float v[4];
    float s1 = 0.0f, s2 = 0.0f;
#pragma unroll
    for (int j = 0; j < 4; ++j) {
        const int c = j * 128 + tid;
        v[j] = bf2f(rp[c]) + bf2f(sp[c]);
        s1 += v[j];
        s2 += v[j] * v[j];
    }
    red[tid] = s1; __syncthreads();
    for (int off = 64; off > 0; off >>= 1) {
        if (tid < off) red[tid] += red[tid + off];
        __syncthreads();
    }
    const float mean = red[0] * (1.0f / (float)DMODEL);
    __syncthreads();
    red[tid] = s2; __syncthreads();
    for (int off = 64; off > 0; off >>= 1) {
        if (tid < off) red[tid] += red[tid + off];
        __syncthreads();
    }
    const float var  = red[0] * (1.0f / (float)DMODEL) - mean * mean;
    const float rstd = rsqrtf(var + 1e-5f);
#pragma unroll
    for (int j = 0; j < 4; ++j) {
        const int c = j * 128 + tid;
        const float r = (v[j] - mean) * rstd * g[c] + be[c];
        if (F32OUT) ((float*)outv)[(size_t)row * DMODEL + c] = r;
        else ((unsigned short*)outv)[(size_t)row * DMODEL + c] = f2bf(r);
    }
}

// ---------------------------------------------------------------------------
// Host orchestration.
// Flattened input order (reference dict insertion order):
//   0 src, 1 tgt, 2 src_mask, 3 tgt_mask, 4 src_emb, 5 tgt_emb,
//   enc layer L (base 6 + 16L): wq,bq,wk,bk,wv,bv,wo,bo, w1,b1,w2,b2,
//                               ln1_g,ln1_b,ln2_g,ln2_b
//   dec layer L (base 102 + 26L): self(8), cross(8), ff(4),
//                               ln1_g,ln1_b,ln2_g,ln2_b,ln3_g,ln3_b
// ---------------------------------------------------------------------------
extern "C" void kernel_launch(void* const* d_in, const int* in_sizes, int n_in,
                              void* d_out, int out_size, void* d_ws, size_t ws_size,
                              hipStream_t stream)
{
    (void)in_sizes; (void)n_in; (void)out_size; (void)ws_size;

    const int*   src     = (const int*)d_in[0];
    const int*   tgt     = (const int*)d_in[1];
    const float* src_emb = (const float*)d_in[4];
    const float* tgt_emb = (const float*)d_in[5];

    auto F = [&](int i) { return (const float*)d_in[i]; };

    const size_t NACT = (size_t)NROWS * DMODEL;   // 2,097,152 elements
    unsigned short* ws   = (unsigned short*)d_ws;
    unsigned short* X    = ws;               // encoder activations / memory
    unsigned short* Y    = ws + 1 * NACT;    // decoder activations
    unsigned short* Q    = ws + 2 * NACT;
    unsigned short* Kb   = ws + 3 * NACT;
    unsigned short* Vb   = ws + 4 * NACT;
    unsigned short* ATT  = ws + 5 * NACT;
    unsigned short* PROJ = ws + 6 * NACT;
    unsigned short* MID  = Q;                // FF hidden reuses Q..ATT (4*NACT)

    auto gemm = [&](const unsigned short* A, const float* W, const float* Bv,
                    unsigned short* C, int N, int K, int relu) {
        dim3 g(N / TN, NROWS / TM);
        gemm_bias_kernel<<<g, 256, 0, stream>>>(A, W, Bv, C, NROWS, N, K, relu);
    };
    auto attn = [&](const unsigned short* q, const unsigned short* k,
                    const unsigned short* v, unsigned short* o, int causal) {
        dim3 g(NS / 128, NH, NB);
        attention_kernel<<<g, 256, 0, stream>>>(q, k, v, o, causal);
    };
    auto ln = [&](const unsigned short* r, const unsigned short* s,
                  int gi, int bi, unsigned short* o) {
        add_ln_kernel<false><<<NROWS, 128, 0, stream>>>(r, s, F(gi), F(bi), (void*)o);
    };

    // -------------------- Encoder --------------------
    embed_kernel<<<NROWS, 128, 0, stream>>>(src, src_emb, X);
    for (int L = 0; L < 6; ++L) {
        const int p = 6 + L * 16;
        gemm(X, F(p + 0), F(p + 1), Q,  DMODEL, DMODEL, 0);   // wq,bq
        gemm(X, F(p + 2), F(p + 3), Kb, DMODEL, DMODEL, 0);   // wk,bk
        gemm(X, F(p + 4), F(p + 5), Vb, DMODEL, DMODEL, 0);   // wv,bv
        attn(Q, Kb, Vb, ATT, 0);
        gemm(ATT, F(p + 6), F(p + 7), PROJ, DMODEL, DMODEL, 0); // wo,bo
        ln(X, PROJ, p + 12, p + 13, X);
        gemm(X,   F(p + 8),  F(p + 9),  MID,  DFF,    DMODEL, 1); // w1,b1 + relu
        gemm(MID, F(p + 10), F(p + 11), PROJ, DMODEL, DFF,    0); // w2,b2
        ln(X, PROJ, p + 14, p + 15, X);
    }
    // X now holds the encoder memory m.

    // -------------------- Decoder --------------------
    embed_kernel<<<NROWS, 128, 0, stream>>>(tgt, tgt_emb, Y);
    for (int L = 0; L < 6; ++L) {
        const int p = 102 + L * 26;
        // self-attention (causal)
        gemm(Y, F(p + 0), F(p + 1), Q,  DMODEL, DMODEL, 0);
        gemm(Y, F(p + 2), F(p + 3), Kb, DMODEL, DMODEL, 0);
        gemm(Y, F(p + 4), F(p + 5), Vb, DMODEL, DMODEL, 0);
        attn(Q, Kb, Vb, ATT, 1);
        gemm(ATT, F(p + 6), F(p + 7), PROJ, DMODEL, DMODEL, 0);
        ln(Y, PROJ, p + 20, p + 21, Y);
        // cross-attention (q from y, k/v from encoder memory X)
        gemm(Y, F(p + 8),  F(p + 9),  Q,  DMODEL, DMODEL, 0);
        gemm(X, F(p + 10), F(p + 11), Kb, DMODEL, DMODEL, 0);
        gemm(X, F(p + 12), F(p + 13), Vb, DMODEL, DMODEL, 0);
        attn(Q, Kb, Vb, ATT, 0);
        gemm(ATT, F(p + 14), F(p + 15), PROJ, DMODEL, DMODEL, 0);
        ln(Y, PROJ, p + 22, p + 23, Y);
        // feed-forward
        gemm(Y,   F(p + 16), F(p + 17), MID,  DFF,    DMODEL, 1);
        gemm(MID, F(p + 18), F(p + 19), PROJ, DMODEL, DFF,    0);
        if (L == 5) {
            add_ln_kernel<true><<<NROWS, 128, 0, stream>>>(
                Y, PROJ, F(p + 24), F(p + 25), d_out);
        } else {
            ln(Y, PROJ, p + 24, p + 25, Y);
        }
    }
}
